// SimpleChemistryDecoder_62912680952406
// MI455X (gfx1250) — matrix operand order repaired and checked by reading, not compiled, not run on hardware
//
#include <hip/hip_runtime.h>
#include <stdint.h>

typedef _Float16 half16 __attribute__((ext_vector_type(16)));
typedef float    float8 __attribute__((ext_vector_type(8)));

#define SD   128
#define CD   768
#define H1S  64
#define H1C  192
#define KT_S (SD / 32)    // 4  K-tiles, structural
#define NT_S (H1S / 16)   // 4  N-tiles, structural
#define KT_C (CD / 32)    // 24 K-tiles, chemical
#define NT_C (H1C / 16)   // 12 N-tiles, chemical
#define SSTR (SD + 8)     // LDS row stride (halfs), bank-conflict pad (8B-aligned)
#define CSTR (CD + 8)
#define MBLK 32           // edges per block = two 16-row WMMA M-tiles
#define W1S_PACK_HALFS (KT_S * NT_S * 512)   // 8192   halfs = 16 KB
#define W1C_PACK_HALFS (KT_C * NT_C * 512)   // 147456 halfs = 288 KB

// ds_swizzle_b32 XOR patterns (group-of-32: and=0x1F, or=0, xor in [14:10])
#define SWZ_X1 0x041F
#define SWZ_X2 0x081F
#define SWZ_X4 0x101F
#define SWZ_X8 0x201F
#define SWZ_ADD(v, pat) \
  ((v) + __int_as_float(__builtin_amdgcn_ds_swizzle(__float_as_int(v), (pat))))

// ---------------------------------------------------------------------------
// Repack W (row-major K x N, fp32) into f16 WMMA B-fragments:
// fragment f = kt*numNT + nt holds 32 lanes x 16 halfs, lane l covers column
// n = nt*16 + (l&15); half e covers k = kt*32 + (e>=8 ? 16 : 0) + (l>=16 ? 8 : 0) + (e&7).
// ---------------------------------------------------------------------------
__global__ void pack_weights_kernel(const float* __restrict__ W1s,
                                    const float* __restrict__ W1c,
                                    _Float16* __restrict__ pack) {
  int idx = blockIdx.x * blockDim.x + threadIdx.x;
  if (idx >= W1S_PACK_HALFS + W1C_PACK_HALFS) return;
  const float* W; int Ntot, numNT, local; _Float16* dst;
  if (idx < W1S_PACK_HALFS) {
    W = W1s; Ntot = H1S; numNT = NT_S; local = idx; dst = pack;
  } else {
    W = W1c; Ntot = H1C; numNT = NT_C; local = idx - W1S_PACK_HALFS;
    dst = pack + W1S_PACK_HALFS;
  }
  int e  = local & 15;
  int l  = (local >> 4) & 31;
  int f  = local >> 9;
  int nt = f % numNT;
  int kt = f / numNT;
  int k  = kt * 32 + ((e >> 3) << 4) + ((l >> 4) << 3) + (e & 7);
  int n  = nt * 16 + (l & 15);
  dst[local] = (_Float16)W[k * Ntot + n];
}

// A fragment (16x32 f16) from LDS per the documented CDNA5 layout.
// Per-lane pattern is two contiguous 8-half runs -> ds_load_b128 pairs.
__device__ __forceinline__ half16 load_a_frag(const _Float16* A, int stride,
                                              int kt, int lane) {
  union { half16 v; uint32_t u[8]; } fr;
  int m   = lane & 15;
  int off = (lane >> 4) << 3;           // 0 or 8
  const _Float16* row = A + m * stride + kt * 32;
#pragma unroll
  for (int p = 0; p < 8; ++p) {
    int k = 2 * p + ((p >= 4) ? 8 : 0) + off;
    fr.u[p] = *(const uint32_t*)(row + k);
  }
  return fr.v;
}

__device__ __forceinline__ half16 load_b_frag(const _Float16* pack, int numNT,
                                              int kt, int nt, int lane) {
  return *(const half16*)(pack + ((((size_t)(kt * numNT + nt)) * 32 + lane) << 4));
}

// bias + ReLU + second-layer column weight, accumulated into per-lane partials.
// C layout: VGPR r -> row r (lanes 0-15) or r+8 (lanes 16-31); col = lane&15.
__device__ __forceinline__ void tile_partials(float8 acc,
                                              const float* __restrict__ b1,
                                              const float* __restrict__ w2,
                                              int nt, int lane, float part[8]) {
  int n      = nt * 16 + (lane & 15);
  float bias = b1[n];
  float w    = w2[n];
#pragma unroll
  for (int r = 0; r < 8; ++r)
    part[r] += fmaxf(acc[r] + bias, 0.f) * w;
}

// 16-column XOR-tree sum within each half-wave via ds_swizzle_b32
__device__ __forceinline__ void reduce16(float part[8]) {
#pragma unroll
  for (int r = 0; r < 8; ++r) {
    part[r] = SWZ_ADD(part[r], SWZ_X1);
    part[r] = SWZ_ADD(part[r], SWZ_X2);
    part[r] = SWZ_ADD(part[r], SWZ_X4);
    part[r] = SWZ_ADD(part[r], SWZ_X8);
  }
}

// product of two float4 rows -> 4 packed f16 -> one 8-byte LDS store
__device__ __forceinline__ void store_h4(_Float16* dst, float4 a, float4 b) {
  union { _Float16 h[4]; uint2 u; } t;
  t.h[0] = (_Float16)(a.x * b.x);
  t.h[1] = (_Float16)(a.y * b.y);
  t.h[2] = (_Float16)(a.z * b.z);
  t.h[3] = (_Float16)(a.w * b.w);
  *(uint2*)dst = t.u;
}

__global__ void __launch_bounds__(256)
edge_decoder_kernel(const float* __restrict__ z,
                    const float* __restrict__ chem,
                    const long long* __restrict__ edge,
                    const float* __restrict__ smiles,
                    const _Float16* __restrict__ pack,
                    const float* __restrict__ b1s,
                    const float* __restrict__ W2s,
                    const float* __restrict__ b2s,
                    const float* __restrict__ b1c,
                    const float* __restrict__ W2c,
                    const float* __restrict__ b2c,
                    const float* __restrict__ alpha_p,
                    float* __restrict__ out, int E) {
  __shared__ _Float16 As[MBLK * SSTR];    //  ~8.7 KB
  __shared__ _Float16 Ac[MBLK * CSTR];    // ~49.7 KB
  __shared__ float s_score_s[MBLK];
  __shared__ float s_score_c[MBLK];
  __shared__ float s_valid[MBLK];
  __shared__ int   s_src[MBLK], s_dst[MBLK];

  const int tid  = threadIdx.x;
  const int lane = tid & 31;
  const int wave = tid >> 5;              // 0..7
  const int base = blockIdx.x * MBLK;

  if (tid < MBLK) {
    int ge = base + tid;
    long long s = 0, d = 0;
    if (ge < E) {
      s = edge[2 * (long long)ge];
      d = edge[2 * (long long)ge + 1];
    }
    s_src[tid]     = (int)s;
    s_dst[tid]     = (int)d;
    s_valid[tid]   = smiles[(int)s] * smiles[(int)d];
    s_score_s[tid] = 0.f;
    s_score_c[tid] = 0.f;
  }
  __syncthreads();

  // Gather node rows as float4, form products, stage as packed f16 in LDS.
  for (int idx = tid; idx < MBLK * (SD / 4); idx += 256) {      // 1024 total
    int m  = idx >> 5;                                          // / (SD/4)
    int c4 = idx & 31;
    float4 a = *(const float4*)(z + (size_t)s_src[m] * SD + c4 * 4);
    float4 b = *(const float4*)(z + (size_t)s_dst[m] * SD + c4 * 4);
    store_h4(&As[m * SSTR + c4 * 4], a, b);
  }
  for (int idx = tid; idx < MBLK * (CD / 4); idx += 256) {      // 6144 total
    int m  = idx / (CD / 4);
    int c4 = idx - m * (CD / 4);
    float4 a = *(const float4*)(chem + (size_t)s_src[m] * CD + c4 * 4);
    float4 b = *(const float4*)(chem + (size_t)s_dst[m] * CD + c4 * 4);
    store_h4(&Ac[m * CSTR + c4 * 4], a, b);
  }
  __syncthreads();

  const _Float16* packS = pack;
  const _Float16* packC = pack + W1S_PACK_HALFS;

  // ---- Structural GEMM: 2 mt x 4 nt = 8 tiles -> 1 per wave --------------
  const int mt_s = wave >> 2;
  const int nt_s = wave & 3;
  float part_s[8] = {0.f, 0.f, 0.f, 0.f, 0.f, 0.f, 0.f, 0.f};
  {
    float8 acc = {};
    const _Float16* Abase = As + mt_s * 16 * SSTR;
    for (int kt = 0; kt < KT_S; ++kt) {
      half16 a = load_a_frag(Abase, SSTR, kt, lane);
      half16 b = load_b_frag(packS, NT_S, kt, nt_s, lane);
      acc = __builtin_amdgcn_wmma_f32_16x16x32_f16(false, a, false, b,
                                                   (short)0, acc, false, false);
    }
    tile_partials(acc, b1s, W2s, nt_s, lane, part_s);
  }

  // ---- Chemical GEMM: 2 mt x 12 nt = 24 tiles -> 3 per wave --------------
  // tile t = wave + 8*i ; mt = t/12, nt = t%12. Partials bucketed by mt.
  float part_c0[8] = {0.f, 0.f, 0.f, 0.f, 0.f, 0.f, 0.f, 0.f};  // mt = 0
  float part_c1[8] = {0.f, 0.f, 0.f, 0.f, 0.f, 0.f, 0.f, 0.f};  // mt = 1
#pragma unroll
  for (int i = 0; i < 3; ++i) {
    const int t  = wave + 8 * i;
    const int mt = t / NT_C;
    const int nt = t % NT_C;
    const _Float16* Abase = Ac + mt * 16 * CSTR;
    float8 acc = {};
    for (int kt = 0; kt < KT_C; ++kt) {
      half16 a = load_a_frag(Abase, CSTR, kt, lane);
      half16 b = load_b_frag(packC, NT_C, kt, nt, lane);
      acc = __builtin_amdgcn_wmma_f32_16x16x32_f16(false, a, false, b,
                                                   (short)0, acc, false, false);
    }
    tile_partials(acc, b1c, W2c, nt, lane, mt == 0 ? part_c0 : part_c1);
  }

  // ---- Column reductions (once per bucket) and cross-wave accumulation ---
  reduce16(part_s);
  reduce16(part_c0);
  reduce16(part_c1);
  if ((lane & 15) == 0) {
    int rowbase = (lane >> 4) << 3;     // lane 0 -> rows 0..7, lane 16 -> 8..15
#pragma unroll
    for (int r = 0; r < 8; ++r) {
      atomicAdd(&s_score_s[mt_s * 16 + rowbase + r], part_s[r]);
      atomicAdd(&s_score_c[rowbase + r],             part_c0[r]);
      atomicAdd(&s_score_c[16 + rowbase + r],        part_c1[r]);
    }
  }
  __syncthreads();

  if (tid < MBLK) {
    int ge = base + tid;
    if (ge < E) {
      float ss = s_score_s[tid] + b2s[0];
      float sc = s_score_c[tid] + b2c[0];
      out[ge] = ss + alpha_p[0] * s_valid[tid] * sc;
    }
  }
}

extern "C" void kernel_launch(void* const* d_in, const int* in_sizes, int n_in,
                              void* d_out, int out_size, void* d_ws, size_t ws_size,
                              hipStream_t stream) {
  const float*     z      = (const float*)d_in[0];
  const float*     chem   = (const float*)d_in[1];
  const long long* edge   = (const long long*)d_in[2];   // int64 per reference
  const float*     smiles = (const float*)d_in[3];
  const float*     W1s    = (const float*)d_in[4];
  const float*     b1s    = (const float*)d_in[5];
  const float*     W2s    = (const float*)d_in[6];
  const float*     b2s    = (const float*)d_in[7];
  const float*     W1c    = (const float*)d_in[8];
  const float*     b1c    = (const float*)d_in[9];
  const float*     W2c    = (const float*)d_in[10];
  const float*     b2c    = (const float*)d_in[11];
  const float*     alpha  = (const float*)d_in[12];
  float*           out    = (float*)d_out;
  _Float16*        pack   = (_Float16*)d_ws;             // 304 KB used

  const int E = in_sizes[2] / 2;

  const int pack_total = W1S_PACK_HALFS + W1C_PACK_HALFS;
  pack_weights_kernel<<<(pack_total + 255) / 256, 256, 0, stream>>>(W1s, W1c, pack);

  const int nblocks = (E + MBLK - 1) / MBLK;
  edge_decoder_kernel<<<nblocks, 256, 0, stream>>>(
      z, chem, edge, smiles, pack, b1s, W2s, b2s, b1c, W2c, b2c, alpha, out, E);
}
